// DualCollaborativeLevelEncoder_33380485824843
// MI455X (gfx1250) — compile-verified, matrix operand order checked by reading, not run on hardware
//
#include <hip/hip_runtime.h>
#include <hip/hip_bf16.h>
#include <math.h>

typedef __bf16 bf16_t;
typedef __attribute__((ext_vector_type(16))) __bf16 v16bf;
typedef __attribute__((ext_vector_type(8)))  __bf16 v8bf;
typedef __attribute__((ext_vector_type(8)))  float  v8f;
typedef __attribute__((ext_vector_type(4)))  unsigned int u32x4;
typedef __attribute__((ext_vector_type(8)))  int i32x8;
typedef __attribute__((ext_vector_type(4)))  int i32x4;

#define BSZ   16
#define NRR   100
#define NGG   196
#define NAA   296
#define DMOD  512
#define NHEAD 8
#define DFF   2048

// ---------------------------------------------------------------- utilities
__device__ __forceinline__ float pos_emb_val(int n, int d, int N) {
  float p = (float)n / ((float)(N - 1) + 1e-6f) * 6.28318530717958647692f;
  float t = __powf(10000.0f, (float)(d & ~1) * (1.0f / (float)DMOD));
  float a = p / t;
  return (d & 1) ? __cosf(a) : __sinf(a);
}

__global__ void k_cvt_bf16(const float* __restrict__ in, bf16_t* __restrict__ out, size_t n) {
  size_t i = (size_t)blockIdx.x * blockDim.x + threadIdx.x;
  size_t stride = (size_t)gridDim.x * blockDim.x;
  for (; i < n; i += stride) out[i] = (bf16_t)in[i];
}

// TDM: issue a 2D tensor-tile load (global -> LDS), D# per CDNA5 ISA ch.8.
// dim0/dim1 = tensor extents (elements), stride0 = row stride (elements),
// tile0 x tile1 = tile shape, g1d0 packs data_size + LDS pad config.
// This toolchain exposes the 6-arg builtin (groups 0..3 + extra + cpol).
__device__ __forceinline__ void tdm_load_2d(const void* gptr, unsigned lds_byte_off,
                                            unsigned dim0, unsigned dim1,
                                            unsigned stride0, unsigned tile0,
                                            unsigned tile1, unsigned g1d0) {
  unsigned long long ga = (unsigned long long)(uintptr_t)gptr;
  u32x4 g0;
  g0[0] = 1u;                                   // count=1 (valid), user mode
  g0[1] = lds_byte_off;                         // lds_addr
  g0[2] = (unsigned)ga;                         // global_addr[31:0]
  g0[3] = (unsigned)(ga >> 32) | 0x80000000u;   // global_addr[56:32] | type=2
  i32x8 g1;
  g1[0] = (int)g1d0;                            // mask=0, data_size, pad cfg
  g1[1] = (int)((dim0 & 0xffffu) << 16);        // tensor_dim0[15:0] @ bit48
  g1[2] = (int)((dim0 >> 16) | ((dim1 & 0xffffu) << 16));
  g1[3] = (int)((dim1 >> 16) | (tile0 << 16));  // tile_dim0
  g1[4] = (int)tile1;                           // tile_dim1 (tile_dim2=0)
  g1[5] = (int)stride0;                         // tensor_dim0_stride[31:0]
  g1[6] = 0;
  g1[7] = 0;
  i32x4 gz4 = {0, 0, 0, 0};
  i32x8 gz8 = {0, 0, 0, 0, 0, 0, 0, 0};
  __builtin_amdgcn_tensor_load_to_lds(g0, g1, gz4, gz4, gz8, 0);
}

// data_size=2B (code 1) | pad_enable | pad_interval=16dw | pad_amount=4dw
// => LDS row stride = 16+4 dwords = 40 bf16 (matches LDA, kills bank conflicts)
#define TDM_G1D0_BF16_PAD16_4 (0x00010000u | 0x00100000u | (3u << 22) | (3u << 25))

// --------------------------------------------------------- geometry helpers
__global__ void k_boxprep(const float* __restrict__ rb, const float* __restrict__ gb,
                          float* __restrict__ bp) {
  int idx = blockIdx.x * blockDim.x + threadIdx.x;  // b*NAA + n
  if (idx >= BSZ * NAA) return;
  int b = idx / NAA, n = idx % NAA;
  const float* s = (n < NRR) ? (rb + ((size_t)b * NRR + n) * 4)
                             : (gb + ((size_t)b * NGG + (n - NRR)) * 4);
  float xm = s[0], ym = s[1], xM = s[2], yM = s[3];
  float* o = bp + (size_t)idx * 4;
  o[0] = (xm + xM) * 0.5f;  o[1] = (ym + yM) * 0.5f;
  o[2] = xM - xm + 1.0f;    o[3] = yM - ym + 1.0f;
}

__global__ void k_mask(const unsigned char* __restrict__ rm, const unsigned char* __restrict__ gm,
                       unsigned char* __restrict__ am) {
  int idx = blockIdx.x * blockDim.x + threadIdx.x;
  if (idx >= BSZ * NAA) return;
  int b = idx / NAA, n = idx % NAA;
  am[idx] = (n < NRR) ? rm[b * NRR + n] : gm[b * NGG + (n - NRR)];
}

// Precompute log(clip(relu(rel . w + b), 1e-6)) bias table [B, H, NAA, NAA]
__global__ void __launch_bounds__(256)
k_relw(const float* __restrict__ bp, const float* __restrict__ fcw,
       const float* __restrict__ fcb, float* __restrict__ LB) {
  __shared__ float sfw[NHEAD * 64];
  __shared__ float sfb[NHEAD];
  int tid = threadIdx.x;
  if (tid < NHEAD * 64) sfw[tid] = fcw[tid];
  if (tid >= 256 - NHEAD) sfb[tid - (256 - NHEAD)] = fcb[tid - (256 - NHEAD)];
  __syncthreads();

  int b = blockIdx.y;
  int idx = blockIdx.x * 256 + tid;
  if (idx >= NAA * NAA) return;
  int i = idx / NAA, j = idx % NAA;
  const float* bi = bp + ((size_t)b * NAA + i) * 4;
  const float* bj = bp + ((size_t)b * NAA + j) * 4;
  float dx = logf(fmaxf(fabsf((bi[0] - bj[0]) / bi[2]), 1e-3f));
  float dy = logf(fmaxf(fabsf((bi[1] - bj[1]) / bi[3]), 1e-3f));
  float dw = logf(bi[2] / bj[2]);
  float dh = logf(bi[3] / bj[3]);
  float pos4[4] = {dx, dy, dw, dh};
  float rel[64];
#pragma unroll
  for (int p = 0; p < 4; ++p) {
#pragma unroll
    for (int f = 0; f < 8; ++f) {
      float ang = 100.0f * pos4[p] * __powf(1000.0f, -(float)f * 0.125f);
      rel[p * 8 + f]      = __sinf(ang);
      rel[32 + p * 8 + f] = __cosf(ang);
    }
  }
#pragma unroll
  for (int h = 0; h < NHEAD; ++h) {
    float acc = sfb[h];
#pragma unroll
    for (int g = 0; g < 64; ++g) acc += rel[g] * sfw[h * 64 + g];
    float w = fmaxf(acc, 0.0f);
    LB[(((size_t)b * NHEAD + h) * NAA + i) * NAA + j] = logf(fmaxf(w, 1e-6f));
  }
}

// -------------------------------------------- LayerNorm (+residual/pe/mask)
__global__ void __launch_bounds__(256)
k_ln512(const float* __restrict__ x, const float* __restrict__ res,
        const float* __restrict__ gamma, const float* __restrict__ beta,
        const unsigned char* __restrict__ mask, int seq, int pe_n,
        float* __restrict__ outf, bf16_t* __restrict__ outb) {
  __shared__ float sred[256];
  int row = blockIdx.x;
  int n = row % seq;
  int tid = threadIdx.x;
  size_t base = (size_t)row * DMOD;
  float v0 = x[base + tid];
  float v1 = x[base + tid + 256];
  if (res) { v0 += res[base + tid]; v1 += res[base + tid + 256]; }
  sred[tid] = v0 + v1; __syncthreads();
  for (int o = 128; o > 0; o >>= 1) { if (tid < o) sred[tid] += sred[tid + o]; __syncthreads(); }
  float mean = sred[0] * (1.0f / (float)DMOD);
  __syncthreads();
  float d0 = v0 - mean, d1 = v1 - mean;
  sred[tid] = d0 * d0 + d1 * d1; __syncthreads();
  for (int o = 128; o > 0; o >>= 1) { if (tid < o) sred[tid] += sred[tid + o]; __syncthreads(); }
  float inv = rsqrtf(sred[0] * (1.0f / (float)DMOD) + 1e-5f);
  float y0 = d0 * inv * gamma[tid]       + beta[tid];
  float y1 = d1 * inv * gamma[tid + 256] + beta[tid + 256];
  if (pe_n) { y0 += pos_emb_val(n, tid, pe_n); y1 += pos_emb_val(n, tid + 256, pe_n); }
  if (mask && mask[row]) { y0 = 0.0f; y1 = 0.0f; }
  outf[base + tid] = y0; outf[base + tid + 256] = y1;
  if (outb) { outb[base + tid] = (bf16_t)y0; outb[base + tid + 256] = (bf16_t)y1; }
}

// comb = concat(reg, grd) + pos_emb(296), bf16 only (feeds K/V GEMM)
__global__ void __launch_bounds__(256)
k_concat_pe(const float* __restrict__ regf, const float* __restrict__ grdf,
            bf16_t* __restrict__ comb) {
  int row = blockIdx.x;  // b*NAA + n
  int b = row / NAA, n = row % NAA;
  int tid = threadIdx.x;
  const float* src = (n < NRR) ? regf + ((size_t)b * NRR + n) * DMOD
                               : grdf + ((size_t)b * NGG + (n - NRR)) * DMOD;
  size_t base = (size_t)row * DMOD;
  float y0 = src[tid]       + pos_emb_val(n, tid, NAA);
  float y1 = src[tid + 256] + pos_emb_val(n, tid + 256, NAA);
  comb[base + tid] = (bf16_t)y0; comb[base + tid + 256] = (bf16_t)y1;
}

// ------------------------------------------------------- WMMA bf16 GEMM
// C[M,N] = act(A[M,K] x B[K,N] + bias); A,B bf16 row-major; out fp32 (+bf16).
// Block: 256 threads = 8 waves; block tile 64x64; wave tile 16x32 (2 WMMA
// sharing one A fragment); K-step 32. Double-buffered LDS pipeline:
//   A tile staged by the Tensor Data Mover (tensor_load_to_lds, TENSORcnt),
//   with TDM pad fields producing the stride-40 (bank-friendly) layout;
//   B tile staged transposed by the waves; ONE barrier per K-step.
#define TM 64
#define TN 64
#define TK 32
#define LDA 40
#define LDBT 40

__global__ void __launch_bounds__(256)
k_gemm_bf16(const bf16_t* __restrict__ A, const bf16_t* __restrict__ Bw,
            const float* __restrict__ bias, float* __restrict__ Cf,
            bf16_t* __restrict__ Cb, int M, int N, int K, int relu) {
  __shared__ bf16_t sA[2][TM * LDA];    // [m][k], stride 40 (TDM-padded)
  __shared__ bf16_t sBt[2][TN * LDBT];  // [n][k] (transposed)
  int tid  = threadIdx.x;
  int lane = tid & 31;
  int wave = tid >> 5;        // 0..7
  int wm = wave & 3;          // 4 row tiles of 16
  int wn = wave >> 2;         // 2 col tiles of 32
  int m0 = blockIdx.y * TM;
  int n0 = blockIdx.x * TN;
  int half = lane >> 4;
  int ml = lane & 15;

  int brow = tid >> 3;             // 0..31 (k)
  int bcol = (tid & 7) * 8;        // 0..56 (n)

  int am  = wm * 16 + ml;
  int bn0 = wn * 32 + ml;
  int bn1 = bn0 + 16;

  const bf16_t* Abase = A + (size_t)m0 * K;
  unsigned ldsA0 = (unsigned)(uintptr_t)&sA[0][0];
  unsigned ldsA1 = (unsigned)(uintptr_t)&sA[1][0];

  // ---- prologue: stage buffer 0
  if (wave == 0)
    tdm_load_2d(Abase, ldsA0, (unsigned)K, (unsigned)(M - m0), (unsigned)K,
                TK, TM, TDM_G1D0_BF16_PAD16_4);
  {
    v8bf bv = *(const v8bf*)(Bw + (size_t)brow * N + n0 + bcol);
#pragma unroll
    for (int i = 0; i < 8; ++i) sBt[0][(bcol + i) * LDBT + brow] = bv[i];
  }
  if (wave == 0) __builtin_amdgcn_s_wait_tensorcnt(0);
  __syncthreads();

  v8f acc0 = {}, acc1 = {};
  int cur = 0;
  for (int k0 = 0; k0 < K; k0 += TK) {
    int nxt = cur ^ 1;
    bool has_next = (k0 + TK) < K;
    if (has_next) {
      if (wave == 0)
        tdm_load_2d(Abase + k0 + TK, nxt ? ldsA1 : ldsA0, (unsigned)K,
                    (unsigned)(M - m0), (unsigned)K, TK, TM, TDM_G1D0_BF16_PAD16_4);
      v8bf bv = *(const v8bf*)(Bw + (size_t)(k0 + TK + brow) * N + n0 + bcol);
#pragma unroll
      for (int i = 0; i < 8; ++i) sBt[nxt][(bcol + i) * LDBT + brow] = bv[i];
      if (k0 + 2 * TK < K)
        __builtin_prefetch(Bw + (size_t)(k0 + 2 * TK + brow) * N + n0 + bcol, 0, 3);
    }

    // A fragment: elems 0-7 at k = half*8.., elems 8-15 at k = 16+half*8..
    const bf16_t* cA = &sA[cur][0];
    const bf16_t* cB = &sBt[cur][0];
    v8bf alo = *(const v8bf*)(cA + am * LDA + half * 8);
    v8bf ahi = *(const v8bf*)(cA + am * LDA + 16 + half * 8);
    v16bf afrag = __builtin_shufflevector(alo, ahi,
        0, 1, 2, 3, 4, 5, 6, 7, 8, 9, 10, 11, 12, 13, 14, 15);
    v8bf b0lo = *(const v8bf*)(cB + bn0 * LDBT + half * 8);
    v8bf b0hi = *(const v8bf*)(cB + bn0 * LDBT + 16 + half * 8);
    v16bf bfrag0 = __builtin_shufflevector(b0lo, b0hi,
        0, 1, 2, 3, 4, 5, 6, 7, 8, 9, 10, 11, 12, 13, 14, 15);
    v8bf b1lo = *(const v8bf*)(cB + bn1 * LDBT + half * 8);
    v8bf b1hi = *(const v8bf*)(cB + bn1 * LDBT + 16 + half * 8);
    v16bf bfrag1 = __builtin_shufflevector(b1lo, b1hi,
        0, 1, 2, 3, 4, 5, 6, 7, 8, 9, 10, 11, 12, 13, 14, 15);

    acc0 = __builtin_amdgcn_wmma_f32_16x16x32_bf16(false, afrag, false, bfrag0,
                                                   (short)0, acc0, false, false);
    acc1 = __builtin_amdgcn_wmma_f32_16x16x32_bf16(false, afrag, false, bfrag1,
                                                   (short)0, acc1, false, false);

    if (has_next && wave == 0) __builtin_amdgcn_s_wait_tensorcnt(0);
    __syncthreads();
    cur = nxt;
  }

  int cn0 = n0 + wn * 32 + ml;
  int cn1 = cn0 + 16;
  float bv0 = bias ? bias[cn0] : 0.0f;
  float bv1 = bias ? bias[cn1] : 0.0f;
#pragma unroll
  for (int r = 0; r < 8; ++r) {
    int cm = m0 + wm * 16 + (half << 3) + r;  // C: VGPR r -> row r (+8 hi half)
    float v0 = acc0[r] + bv0;
    float v1 = acc1[r] + bv1;
    if (relu) { v0 = fmaxf(v0, 0.0f); v1 = fmaxf(v1, 0.0f); }
    Cf[(size_t)cm * N + cn0] = v0;
    Cf[(size_t)cm * N + cn1] = v1;
    if (Cb) {
      Cb[(size_t)cm * N + cn0] = (bf16_t)v0;
      Cb[(size_t)cm * N + cn1] = (bf16_t)v1;
    }
  }
}

// -------------------------------------------------- attention (softmax core)
// One wave per (b, h, q) row. Query row staged in LDS (scaled); lanes process
// keys in parallel (k = lane + 32*t) with float4 K-row loads; softmax in LDS.
__global__ void __launch_bounds__(128)
k_attn(const float* __restrict__ Q, const float* __restrict__ K,
       const float* __restrict__ V, const float* __restrict__ LB,
       const unsigned char* __restrict__ am,
       float* __restrict__ Of, bf16_t* __restrict__ Ob,
       int nq, int nk, int qoff, int koff) {
  __shared__ float sq[4][64];
  __shared__ float sw[4][304];
  int bh = blockIdx.x;
  int b = bh >> 3, h = bh & 7;
  int wave = threadIdx.x >> 5, lane = threadIdx.x & 31;
  int q = blockIdx.y * 4 + wave;
  if (q >= nq) return;
  const float scale = 0.125f;  // 1/sqrt(64)
  const float* qrow = Q + ((size_t)b * nq + q) * DMOD + h * 64;
  sq[wave][2 * lane]     = qrow[2 * lane] * scale;
  sq[wave][2 * lane + 1] = qrow[2 * lane + 1] * scale;
  const float* kb = K + (size_t)b * nk * DMOD + h * 64;
  const float* lb = LB + (((size_t)b * NHEAD + h) * NAA + (qoff + q)) * NAA + koff;
  const unsigned char* mk = am + (size_t)b * NAA + koff;

  float mx = -3.0e38f;
  for (int k = lane; k < nk; k += 32) {
    const float* kr = kb + (size_t)k * DMOD;
    float acc = 0.0f;
#pragma unroll
    for (int d = 0; d < 64; d += 4) {
      float4 kv = *(const float4*)(kr + d);
      acc += sq[wave][d] * kv.x + sq[wave][d + 1] * kv.y
           + sq[wave][d + 2] * kv.z + sq[wave][d + 3] * kv.w;
    }
    float logit = acc + lb[k];
    if (mk[k]) logit = -1e9f;
    sw[wave][k] = logit;
    mx = fmaxf(mx, logit);
  }
#pragma unroll
  for (int s = 16; s > 0; s >>= 1) mx = fmaxf(mx, __shfl_xor(mx, s, 32));
  float sum = 0.0f;
  for (int k = lane; k < nk; k += 32) {
    float e = __expf(sw[wave][k] - mx);
    sw[wave][k] = e;
    sum += e;
  }
#pragma unroll
  for (int s = 16; s > 0; s >>= 1) sum += __shfl_xor(sum, s, 32);
  float inv = 1.0f / sum;

  float o0 = 0.0f, o1 = 0.0f;
  const float* vb = V + (size_t)b * nk * DMOD + h * 64;
  for (int k = 0; k < nk; ++k) {
    float w = sw[wave][k] * inv;
    float2 vv = *(const float2*)(vb + (size_t)k * DMOD + 2 * lane);
    o0 += w * vv.x;
    o1 += w * vv.y;
  }
  size_t oi = ((size_t)b * nq + q) * DMOD + h * 64 + 2 * lane;
  Of[oi] = o0; Of[oi + 1] = o1;
  Ob[oi] = (bf16_t)o0; Ob[oi + 1] = (bf16_t)o1;
}

// ------------------------------------------------------------- finalization
__global__ void __launch_bounds__(256)
k_finalize(const float* __restrict__ regf, const float* __restrict__ grdf,
           const unsigned char* __restrict__ am, float* __restrict__ out) {
  int row = blockIdx.x;  // b*NAA + n
  int b = row / NAA, n = row % NAA;
  int tid = threadIdx.x;
  const float* src = (n < NRR) ? regf + ((size_t)b * NRR + n) * DMOD
                               : grdf + ((size_t)b * NGG + (n - NRR)) * DMOD;
  size_t base = (size_t)row * DMOD;
  out[base + tid] = src[tid];
  out[base + tid + 256] = src[tid + 256];
  if (tid == 0) out[(size_t)BSZ * NAA * DMOD + row] = am[row] ? 1.0f : 0.0f;
}

// ------------------------------------------------------------------- driver
static inline size_t align256(size_t x) { return (x + 255) & ~(size_t)255; }

extern "C" void kernel_launch(void* const* d_in, const int* in_sizes, int n_in,
                              void* d_out, int out_size, void* d_ws, size_t ws_size,
                              hipStream_t stream) {
  (void)in_sizes; (void)n_in; (void)out_size; (void)ws_size;
  const float* region_features = (const float*)d_in[0];
  const float* grid_features   = (const float*)d_in[1];
  const float* region_boxes    = (const float*)d_in[2];
  const float* grid_boxes      = (const float*)d_in[3];
  const unsigned char* rm      = (const unsigned char*)d_in[4];
  const unsigned char* gm      = (const unsigned char*)d_in[5];
  const float* attn_W          = (const float*)d_in[6];
  const float* attn_b          = (const float*)d_in[7];
  const float* ln_attn         = (const float*)d_in[8];
  const float* ffn_W1          = (const float*)d_in[9];
  const float* ffn_b1          = (const float*)d_in[10];
  const float* ffn_W2          = (const float*)d_in[11];
  const float* ffn_b2          = (const float*)d_in[12];
  const float* ln_ffn          = (const float*)d_in[13];
  const float* fc_g_w          = (const float*)d_in[14];
  const float* fc_g_b          = (const float*)d_in[15];
  const float* ln_in           = (const float*)d_in[16];

  // deterministic bump allocator over d_ws
  char* wsp = (char*)d_ws;
  size_t off = 0;
  auto alloc = [&](size_t bytes) -> void* {
    void* p = wsp + off; off = align256(off + bytes); return p;
  };
  float*   LB    = (float*)alloc((size_t)BSZ * NHEAD * NAA * NAA * 4);
  float*   bp    = (float*)alloc((size_t)BSZ * NAA * 4 * 4);
  unsigned char* am = (unsigned char*)alloc((size_t)BSZ * NAA);
  float*   regf  = (float*)alloc((size_t)BSZ * NRR * DMOD * 4);
  bf16_t*  regb  = (bf16_t*)alloc((size_t)BSZ * NRR * DMOD * 2);
  float*   grdf  = (float*)alloc((size_t)BSZ * NGG * DMOD * 4);
  bf16_t*  grdb  = (bf16_t*)alloc((size_t)BSZ * NGG * DMOD * 2);
  bf16_t*  combb = (bf16_t*)alloc((size_t)BSZ * NAA * DMOD * 2);
  float*   Qf    = (float*)alloc((size_t)BSZ * NGG * DMOD * 4);
  float*   Kf    = (float*)alloc((size_t)BSZ * NAA * DMOD * 4);
  float*   Vf    = (float*)alloc((size_t)BSZ * NAA * DMOD * 4);
  float*   AOf   = (float*)alloc((size_t)BSZ * NGG * DMOD * 4);
  bf16_t*  AOb   = (bf16_t*)alloc((size_t)BSZ * NGG * DMOD * 2);
  float*   Opf   = (float*)alloc((size_t)BSZ * NGG * DMOD * 4);
  float*   attf  = (float*)alloc((size_t)BSZ * NGG * DMOD * 4);
  bf16_t*  attb  = (bf16_t*)alloc((size_t)BSZ * NGG * DMOD * 2);
  float*   Hf    = (float*)alloc((size_t)BSZ * NGG * DFF * 4);
  bf16_t*  Hb    = (bf16_t*)alloc((size_t)BSZ * NGG * DFF * 2);
  float*   Ff    = (float*)alloc((size_t)BSZ * NGG * DMOD * 4);
  bf16_t*  Wqkvo = (bf16_t*)alloc((size_t)4 * 3 * 4 * DMOD * DMOD * 2);
  bf16_t*  W1b   = (bf16_t*)alloc((size_t)4 * 3 * DMOD * DFF * 2);
  bf16_t*  W2b   = (bf16_t*)alloc((size_t)4 * 3 * DFF * DMOD * 2);

  // weight conversion (fp32 -> bf16)
  k_cvt_bf16<<<1024, 256, 0, stream>>>(attn_W, Wqkvo, (size_t)4 * 3 * 4 * DMOD * DMOD);
  k_cvt_bf16<<<1024, 256, 0, stream>>>(ffn_W1, W1b,   (size_t)4 * 3 * DMOD * DFF);
  k_cvt_bf16<<<1024, 256, 0, stream>>>(ffn_W2, W2b,   (size_t)4 * 3 * DFF * DMOD);

  // geometry bias + masks
  k_boxprep<<<(BSZ * NAA + 255) / 256, 256, 0, stream>>>(region_boxes, grid_boxes, bp);
  k_mask<<<(BSZ * NAA + 255) / 256, 256, 0, stream>>>(rm, gm, am);
  { dim3 rg((NAA * NAA + 255) / 256, BSZ);
    k_relw<<<rg, 256, 0, stream>>>(bp, fc_g_w, fc_g_b, LB); }

  // input LN + positional embedding
  k_ln512<<<BSZ * NRR, 256, 0, stream>>>(region_features, nullptr, ln_in + 0, ln_in + 512,
                                         nullptr, NRR, NRR, regf, regb);
  k_ln512<<<BSZ * NGG, 256, 0, stream>>>(grid_features, nullptr, ln_in + 1024, ln_in + 1536,
                                         nullptr, NGG, NGG, grdf, grdb);

  auto gemm = [&](const bf16_t* A, const bf16_t* Bw, const float* bias,
                  float* Cf, bf16_t* Cb, int M, int N, int K, int relu) {
    dim3 g(N / TN, M / TM);
    k_gemm_bf16<<<g, 256, 0, stream>>>(A, Bw, bias, Cf, Cb, M, N, K, relu);
  };

  auto sub = [&](int type, int l, float* xf, bf16_t* xb, const bf16_t* kvb,
                 int nq, int nk, int qoff, int koff, const unsigned char* qmask) {
    size_t wi = (size_t)(type * 3 + l);
    const bf16_t* WQ = Wqkvo + (wi * 4 + 0) * DMOD * DMOD;
    const bf16_t* WK = Wqkvo + (wi * 4 + 1) * DMOD * DMOD;
    const bf16_t* WV = Wqkvo + (wi * 4 + 2) * DMOD * DMOD;
    const bf16_t* WO = Wqkvo + (wi * 4 + 3) * DMOD * DMOD;
    const float* bQ = attn_b + (wi * 4 + 0) * DMOD;
    const float* bK = attn_b + (wi * 4 + 1) * DMOD;
    const float* bV = attn_b + (wi * 4 + 2) * DMOD;
    const float* bO = attn_b + (wi * 4 + 3) * DMOD;

    gemm(xb,  WQ, bQ, Qf, nullptr, BSZ * nq, DMOD, DMOD, 0);
    gemm(kvb, WK, bK, Kf, nullptr, BSZ * nk, DMOD, DMOD, 0);
    gemm(kvb, WV, bV, Vf, nullptr, BSZ * nk, DMOD, DMOD, 0);

    { dim3 ga(BSZ * NHEAD, (nq + 3) / 4);
      k_attn<<<ga, 128, 0, stream>>>(Qf, Kf, Vf, LB, am, AOf, AOb, nq, nk, qoff, koff); }

    gemm(AOb, WO, bO, Opf, nullptr, BSZ * nq, DMOD, DMOD, 0);
    const float* ga0 = ln_attn + (wi * 2) * DMOD;
    k_ln512<<<BSZ * nq, 256, 0, stream>>>(Opf, xf, ga0, ga0 + DMOD, nullptr, nq, 0, attf, attb);

    const bf16_t* W1 = W1b + wi * DMOD * DFF;
    const bf16_t* W2 = W2b + wi * DFF * DMOD;
    const float* b1 = ffn_b1 + wi * DFF;
    const float* b2 = ffn_b2 + wi * DMOD;
    gemm(attb, W1, b1, Hf, Hb, BSZ * nq, DFF, DMOD, 1);
    gemm(Hb,   W2, b2, Ff, nullptr, BSZ * nq, DMOD, DFF, 0);
    const float* gf0 = ln_ffn + (wi * 2) * DMOD;
    k_ln512<<<BSZ * nq, 256, 0, stream>>>(Ff, attf, gf0, gf0 + DMOD, qmask, nq, 0, xf, xb);
  };

  for (int l = 0; l < 3; ++l) {
    sub(0, l, regf, regb, regb, NRR, NRR, 0,   0,   rm);
    sub(1, l, grdf, grdb, grdb, NGG, NGG, NRR, NRR, gm);
    k_concat_pe<<<BSZ * NAA, 256, 0, stream>>>(regf, grdf, combb);
    sub(2, l, regf, regb, combb, NRR, NAA, 0,   0, rm);
    sub(3, l, grdf, grdb, combb, NGG, NAA, NRR, 0, gm);
  }

  k_finalize<<<BSZ * NAA, 256, 0, stream>>>(regf, grdf, am, (float*)d_out);
}
